// MoveNet_53240414601218
// MI455X (gfx1250) — matrix operand order repaired and checked by reading, not compiled, not run on hardware
//
#include <hip/hip_runtime.h>
#include <hip/hip_bf16.h>

#define FT    48
#define NPIX  (FT * FT)      // 2304
#define CIN   24
#define HC    96
#define SCALE 0.02083333395421505f
#define DELTA 1.8f

typedef __attribute__((ext_vector_type(16))) _Float16 v16h;
typedef __attribute__((ext_vector_type(8)))  float    v8f;

union V16H { v16h v; _Float16 e[16]; };

// ---------------------------------------------------------------------------
// Async global->LDS copy path (CDNA5 GLOBAL_LOAD_ASYNC_TO_LDS_B32, ASYNCcnt)
// Builtin signature (per hipcc diagnostic): (AS1 int*, AS3 int*, imm, imm)
// ---------------------------------------------------------------------------
#if __has_builtin(__builtin_amdgcn_global_load_async_to_lds_b32) && \
    __has_builtin(__builtin_amdgcn_s_wait_asynccnt)
#define HAVE_ASYNC_LDS 1
#else
#define HAVE_ASYNC_LDS 0
#endif

#if HAVE_ASYNC_LDS
typedef __attribute__((address_space(1))) int gint_t;
typedef __attribute__((address_space(3))) int lint_t;
__device__ __forceinline__ gint_t* as_global_(const void* p) {
    // generic global pointer is bit-identical to AS1
    return (gint_t*)(unsigned long long)p;
}
__device__ __forceinline__ lint_t* as_local_(const void* p) {
    // ISA 10.2: LDS address is the low 32 bits of the generic LDS address
    return (lint_t*)(unsigned int)(unsigned long long)p;
}
#endif

// ---------------------------------------------------------------------------
// Kernarg-passed parameter block: all four heads in one launch.
// ---------------------------------------------------------------------------
struct HeadParams {
    const float* dww; const float* dwb;
    const float* w1;  const float* b1;
    const float* w2;  const float* b2;
    float*       outp;                   // [cls][2304] fp32 in workspace
    int          cls;
};
struct AllParams {
    HeadParams h[4];
    const float* x;
};

// ---------------------------------------------------------------------------
// All heads: depthwise3x3+bias -> [WMMA] 24->96 +bias+ReLU -> [WMMA] 96->cls
// grid = 4 heads x 144 N-tiles (16 pixels each), block = 32 threads (wave32).
// ---------------------------------------------------------------------------
__global__ __launch_bounds__(32)
void movenet_heads_kernel(AllParams P)
{
    const int head = blockIdx.x / 144;
    const int tile = blockIdx.x % 144;
    const int lane = threadIdx.x;        // 0..31
    const int p0   = tile * 16;
    const int nn   = lane & 15;          // N within tile / M row for A-frag
    const int hi   = lane >> 4;          // lane half

    const HeadParams hp = P.h[head];
    const float* __restrict__ x    = P.x;
    const float* __restrict__ dww  = hp.dww;
    const float* __restrict__ dwb  = hp.dwb;
    const float* __restrict__ w1   = hp.w1;
    const float* __restrict__ b1   = hp.b1;
    const float* __restrict__ w2   = hp.w2;
    const float* __restrict__ b2   = hp.b2;
    float*       __restrict__ outp = hp.outp;
    const int    cls               = hp.cls;

    __builtin_prefetch(w1, 0, 0);        // global_prefetch_b8
    __builtin_prefetch(w2, 0, 0);

    __shared__ float    dwwsh[CIN * 9];  // depthwise weights (async-staged)
    __shared__ float    dwbsh[CIN];      // depthwise bias    (async-staged)
    __shared__ _Float16 ysh[32][16];     // depthwise out: [K(pad 32)][N]
    __shared__ _Float16 ash[16][32];     // A-tile staging: [M][K]
    __shared__ _Float16 zsh[HC][16];     // PW1 out (relu): [K=96][N]

    // ---- stage depthwise weights into LDS (async-to-LDS when available) ---
#if HAVE_ASYNC_LDS
    for (int i = lane; i < CIN * 9; i += 32)
        __builtin_amdgcn_global_load_async_to_lds_b32(
            as_global_(dww + i), as_local_(&dwwsh[i]), 0, 0);
    if (lane < CIN)
        __builtin_amdgcn_global_load_async_to_lds_b32(
            as_global_(dwb + lane), as_local_(&dwbsh[lane]), 0, 0);
    __builtin_amdgcn_s_wait_asynccnt(0);
#else
    for (int i = lane; i < CIN * 9; i += 32) dwwsh[i] = dww[i];
    if (lane < CIN) dwbsh[lane] = dwb[lane];
#endif
    __syncthreads();

    // ---- depthwise 3x3 + bias for 16 pixels x 24 channels -----------------
    for (int idx = lane; idx < CIN * 16; idx += 32) {
        int c = idx >> 4;
        int i = idx & 15;
        int p = p0 + i, py = p / FT, px = p % FT;
        float acc = dwbsh[c];
        const float* xc = x + c * NPIX;
        const float* wc = dwwsh + c * 9;
        #pragma unroll
        for (int dy = -1; dy <= 1; ++dy) {
            int yy = py + dy; if (yy < 0 || yy >= FT) continue;
            #pragma unroll
            for (int dx = -1; dx <= 1; ++dx) {
                int xx = px + dx; if (xx < 0 || xx >= FT) continue;
                float v = xc[yy * FT + xx] * 0.007843137718737125f - 1.0f;
                acc += v * wc[(dy + 1) * 3 + (dx + 1)];
            }
        }
        ysh[c][i] = (_Float16)acc;
    }
    for (int idx = lane; idx < 8 * 16; idx += 32)            // K pad 24..31 = 0
        ysh[CIN + (idx >> 4)][idx & 15] = (_Float16)0.f;
    __syncthreads();

    // ---- B fragment for PW1: lane = N + 16*(K>=16), h = K%16 --------------
    V16H bfrag;
    #pragma unroll
    for (int h = 0; h < 16; ++h) bfrag.e[h] = ysh[h + 16 * hi][nn];

    // ---- PW1: z[96][16] = W1[96x32] * y[32x16], bias + relu ---------------
    for (int m = 0; m < HC / 16; ++m) {
        for (int idx = lane; idx < 16 * 32; idx += 32) {
            int r = idx >> 5, k = idx & 31;
            ash[r][k] = (k < CIN) ? (_Float16)w1[(16 * m + r) * CIN + k]
                                  : (_Float16)0.f;
        }
        __syncthreads();
        V16H afrag;   // A layout: M=nn; K = h+16*hi
        #pragma unroll
        for (int h = 0; h < 16; ++h) {
            int k = h + 16 * hi;
            afrag.e[h] = ash[nn][k];
        }
        v8f acc = {};
        acc = __builtin_amdgcn_wmma_f32_16x16x32_f16(
                  false, afrag.v, false, bfrag.v, (short)0, acc, false, false);
        #pragma unroll
        for (int i = 0; i < 8; ++i) {   // D layout: M = i + 8*hi, N = nn
            int Mr = i + 8 * hi;
            float v = acc[i] + b1[16 * m + Mr];
            zsh[16 * m + Mr][nn] = (_Float16)(v > 0.f ? v : 0.f);
        }
        __syncthreads();
    }

    // ---- B fragments for PW2 (K = 96 = 3 chunks of 32) --------------------
    V16H b2f[3];
    #pragma unroll
    for (int kc = 0; kc < 3; ++kc)
        #pragma unroll
        for (int h = 0; h < 16; ++h)
            b2f[kc].e[h] = zsh[32 * kc + h + 16 * hi][nn];

    // ---- PW2: out[cls][16] = W2[cls x 96] * z[96x16] + bias ---------------
    const int mt_n = (cls + 15) / 16;
    for (int mt = 0; mt < mt_n; ++mt) {
        v8f acc = {};
        for (int kc = 0; kc < 3; ++kc) {
            for (int idx = lane; idx < 16 * 32; idx += 32) {
                int r = idx >> 5, k = idx & 31;
                int row = 16 * mt + r;
                ash[r][k] = (row < cls) ? (_Float16)w2[row * HC + 32 * kc + k]
                                        : (_Float16)0.f;
            }
            __syncthreads();
            V16H afrag;
            #pragma unroll
            for (int h = 0; h < 16; ++h) {
                int k = h + 16 * hi;
                afrag.e[h] = ash[nn][k];
            }
            acc = __builtin_amdgcn_wmma_f32_16x16x32_f16(
                      false, afrag.v, false, b2f[kc].v, (short)0, acc, false, false);
            __syncthreads();
        }
        #pragma unroll
        for (int i = 0; i < 8; ++i) {
            int Mr  = i + 8 * hi;
            int row = 16 * mt + Mr;
            if (row < cls)
                outp[row * NPIX + p0 + nn] = acc[i] + b2[row];
        }
    }
}

// ---------------------------------------------------------------------------
// Decode: center-weighted argmax + 17 distance-weighted argmaxes -> [17][3]
// ---------------------------------------------------------------------------
__device__ __forceinline__ float sigmoidf_(float v) {
    return 1.0f / (1.0f + __expf(-v));
}

__global__ __launch_bounds__(256)
void movenet_decode_kernel(const float* __restrict__ hm_hp,
                           const float* __restrict__ hm,
                           const float* __restrict__ hps,
                           const float* __restrict__ hpo,
                           float* __restrict__ out)
{
    __shared__ float rv[256];
    __shared__ int   ri[256];
    __shared__ float coor_y[17], coor_x[17];
    __shared__ int   s_ct;

    const int tid = threadIdx.x;

    // --- center-weighted argmax -------------------------------------------
    float best = -1e30f; int bi = 0;
    for (int p = tid; p < NPIX; p += 256) {
        int y = p / FT, xq = p % FT;
        float gy = (float)y - FT * 0.5f, gx = (float)xq - FT * 0.5f;
        float w = 1.0f / (sqrtf(gy * gy + gx * gx) + DELTA);
        float s = sigmoidf_(hm[p]) * w;
        if (s > best) { best = s; bi = p; }
    }
    rv[tid] = best; ri[tid] = bi;
    __syncthreads();
    for (int off = 128; off > 0; off >>= 1) {
        if (tid < off) {
            float ov = rv[tid + off]; int oi = ri[tid + off];
            if (ov > rv[tid] || (ov == rv[tid] && oi < ri[tid])) {
                rv[tid] = ov; ri[tid] = oi;
            }
        }
        __syncthreads();
    }
    if (tid == 0) s_ct = ri[0];
    __syncthreads();

    const int   ct   = s_ct;
    const float ct_y = (float)(ct / FT);
    const float ct_x = (float)(ct % FT);
    if (tid < 17) {
        coor_y[tid] = hps[(2 * tid)     * NPIX + ct] + ct_y;
        coor_x[tid] = hps[(2 * tid + 1) * NPIX + ct] + ct_x;
    }
    __syncthreads();

    // --- per-keypoint distance-weighted argmax ----------------------------
    for (int k = 0; k < 17; ++k) {
        const float ky = coor_y[k], kx = coor_x[k];
        float bv = -1e30f; int bidx = 0;
        for (int p = tid; p < NPIX; p += 256) {
            int y = p / FT, xq = p % FT;
            float dy = (float)y - ky, dx = (float)xq - kx;
            float dw = sqrtf(dy * dy + dx * dx) + DELTA;
            float s  = sigmoidf_(hm_hp[k * NPIX + p]) / dw;
            if (s > bv) { bv = s; bidx = p; }
        }
        rv[tid] = bv; ri[tid] = bidx;
        __syncthreads();
        for (int off = 128; off > 0; off >>= 1) {
            if (tid < off) {
                float ov = rv[tid + off]; int oi = ri[tid + off];
                if (ov > rv[tid] || (ov == rv[tid] && oi < ri[tid])) {
                    rv[tid] = ov; ri[tid] = oi;
                }
            }
            __syncthreads();
        }
        if (tid == 0) {
            int top = ri[0];
            float ys = (float)(top / FT), xs = (float)(top % FT);
            float conf = sigmoidf_(hm_hp[k * NPIX + top]);
            float oy = hpo[(2 * k)     * NPIX + top];
            float ox = hpo[(2 * k + 1) * NPIX + top];
            out[k * 3 + 0] = (oy + ys) * SCALE;
            out[k * 3 + 1] = (ox + xs) * SCALE;
            out[k * 3 + 2] = conf;
        }
        __syncthreads();
    }
}

// ---------------------------------------------------------------------------
extern "C" void kernel_launch(void* const* d_in, const int* in_sizes, int n_in,
                              void* d_out, int out_size, void* d_ws, size_t ws_size,
                              hipStream_t stream)
{
    (void)in_sizes; (void)n_in; (void)out_size; (void)ws_size;

    const float* x  = (const float*)d_in[0];
    float*       ws = (float*)d_ws;

    // heads in setup_inputs() dict order: hm_hp(17), hm(1), hps(34), hp_offset(34)
    const int cls[4] = {17, 1, 34, 34};
    const int off[4] = {0, 17 * NPIX, 18 * NPIX, 52 * NPIX};  // total 86*2304 floats

    AllParams P;
    P.x = x;
    for (int h = 0; h < 4; ++h) {
        P.h[h].dww  = (const float*)d_in[1 + 6 * h + 0];
        P.h[h].dwb  = (const float*)d_in[1 + 6 * h + 1];
        P.h[h].w1   = (const float*)d_in[1 + 6 * h + 2];
        P.h[h].b1   = (const float*)d_in[1 + 6 * h + 3];
        P.h[h].w2   = (const float*)d_in[1 + 6 * h + 4];
        P.h[h].b2   = (const float*)d_in[1 + 6 * h + 5];
        P.h[h].outp = ws + off[h];
        P.h[h].cls  = cls[h];
    }

    movenet_heads_kernel<<<4 * (NPIX / 16), 32, 0, stream>>>(P);

    movenet_decode_kernel<<<1, 256, 0, stream>>>(
        ws + off[0], ws + off[1], ws + off[2], ws + off[3], (float*)d_out);
}
